// mLSTMWeavingCell_42159398978080
// MI455X (gfx1250) — matrix operand order repaired and checked by reading, not compiled, not run on hardware
//
#include <hip/hip_runtime.h>
#include <hip/hip_bf16.h>
#include <math.h>

// Problem constants (match reference)
#define B_   4
#define S_   2048
#define D_   1024
#define NH_  8
#define DH_  128
#define L_   32                  // chunk length (matches WMMA K=32)
#define NCH_ (S_ / L_)           // 64 chunks
#define SCALE_ 0.0883883476483184f   // 1/sqrt(128)
#define EPS_ 1e-6f

typedef __bf16 bf16;
typedef __attribute__((ext_vector_type(16))) bf16  v16bf;
typedef __attribute__((ext_vector_type(8)))  float v8f;

#define WMMA_BF16(a, bm, c) \
  __builtin_amdgcn_wmma_f32_16x16x32_bf16(false, (a), false, (bm), (short)0, (c), false, false)

__device__ __forceinline__ float log_sigmoid_f(float x) {
  return (x >= 0.f) ? -log1pf(__expf(-x)) : (x - log1pf(__expf(x)));
}

// A-operand (16x32 bf16): lane holds row m; K elements are one contiguous
// 16-element run at [half*16] (matches B operand order).
__device__ __forceinline__ v16bf frag_a8(const bf16* base, int half) {
  v16bf r;
  const bf16* p0 = base + half * 16;
  const bf16* p1 = base + half * 16 + 8;
#pragma unroll
  for (int e = 0; e < 8; ++e) { r[e] = p0[e]; r[8 + e] = p1[e]; }
  return r;
}
// B-operand (32x16 bf16): lane holds col n; K elements are one contiguous
// 16-element run at [half*16] (dense analogue of the documented sparse-B layout).
__device__ __forceinline__ v16bf frag_b16(const bf16* p) {
  v16bf r;
#pragma unroll
  for (int e = 0; e < 16; ++e) r[e] = p[e];
  return r;
}

// ---------------------------------------------------------------------------
// Kernel 1: gate pre-activations  ig/fg[b][h][s] = concat(q,k,v)[b,s,:] @ W + b
// grid = B*S blocks, 128 threads.
// ---------------------------------------------------------------------------
__global__ void mlstm_gates_kernel(const float* __restrict__ q,
                                   const float* __restrict__ k,
                                   const float* __restrict__ v,
                                   const float* __restrict__ W_i,
                                   const float* __restrict__ b_i,
                                   const float* __restrict__ W_f,
                                   const float* __restrict__ b_f,
                                   float* __restrict__ ig_ws,
                                   float* __restrict__ fg_ws) {
  __shared__ float red[16][128];
  const int bs  = blockIdx.x;           // b*S + s
  const int tid = threadIdx.x;
  const float* qr = q + (size_t)bs * D_;
  const float* kr = k + (size_t)bs * D_;
  const float* vr = v + (size_t)bs * D_;

  float ai[8], af[8];
#pragma unroll
  for (int hh = 0; hh < 8; ++hh) { ai[hh] = 0.f; af[hh] = 0.f; }

  for (int j = tid; j < D_; j += 128) {
    const float qv = qr[j], kv = kr[j], vv = vr[j];
    const float* wi0 = W_i + (size_t)j * NH_;
    const float* wi1 = W_i + (size_t)(D_ + j) * NH_;
    const float* wi2 = W_i + (size_t)(2 * D_ + j) * NH_;
    const float* wf0 = W_f + (size_t)j * NH_;
    const float* wf1 = W_f + (size_t)(D_ + j) * NH_;
    const float* wf2 = W_f + (size_t)(2 * D_ + j) * NH_;
#pragma unroll
    for (int hh = 0; hh < 8; ++hh) {
      ai[hh] += qv * wi0[hh] + kv * wi1[hh] + vv * wi2[hh];
      af[hh] += qv * wf0[hh] + kv * wf1[hh] + vv * wf2[hh];
    }
  }
#pragma unroll
  for (int hh = 0; hh < 8; ++hh) {
    red[hh][tid]     = ai[hh];
    red[8 + hh][tid] = af[hh];
  }
  __syncthreads();
  if (tid < 16) {
    float s = 0.f;
    for (int l = 0; l < 128; ++l) s += red[tid][l];
    const int b   = bs / S_;
    const int sdx = bs % S_;
    if (tid < 8) {
      ig_ws[((size_t)(b * NH_ + tid)) * S_ + sdx] = s + b_i[tid];
    } else {
      const int hh = tid - 8;
      fg_ws[((size_t)(b * NH_ + hh)) * S_ + sdx] = s + b_f[hh];
    }
  }
}

// ---------------------------------------------------------------------------
// Kernel 2: chunkwise-parallel mLSTM recurrence. One block per (b,h) chain.
// 256 threads = 8 wave32. State C kept as f32 WMMA accumulators (wave w owns
// rows [16w,16w+16)); bf16 mirror in LDS (transposed) feeds Q@C WMMAs.
// ---------------------------------------------------------------------------
__global__ void __launch_bounds__(256)
mlstm_chunk_kernel(const float* __restrict__ q,
                   const float* __restrict__ k,
                   const float* __restrict__ v,
                   const float* __restrict__ c0_in,
                   const float* __restrict__ n0_in,
                   const float* __restrict__ m0_in,
                   const float* __restrict__ ln_w,
                   const float* __restrict__ ig_ws,
                   const float* __restrict__ fg_ws,
                   float* __restrict__ out_h,
                   float* __restrict__ out_c,
                   float* __restrict__ out_n,
                   float* __restrict__ out_m) {
  // LDS
  __shared__ __align__(16) bf16 Ct_lds[DH_ * DH_];   // C^T mirror: [d2][d1], bf16
  __shared__ __align__(16) bf16 Q_lds[L_ * DH_];     // [t][d]
  __shared__ __align__(16) bf16 K_lds[L_ * DH_];     // [t][d]  (pre-scaled k)
  __shared__ __align__(16) bf16 Kw_lds[DH_ * L_];    // [d][t]  (w_t * k_sc)
  __shared__ __align__(16) bf16 Vt_lds[DH_ * L_];    // [d2][t]
  __shared__ __align__(16) bf16 SD_lds[L_ * L_];     // masked decay*S, [t][u]
  __shared__ float Hbuf[L_ * DH_];                   // pre-LN outputs
  __shared__ float n_lds[DH_];
  __shared__ float i_s[L_], f_s[L_], A_s[L_], m_s[L_];
  __shared__ float rowinter_s[L_], wstate_s[L_], denom_s[L_];
  __shared__ float qred[L_][4];
  __shared__ float lnsum[L_][8], lnsq[L_][8], mu_s[L_], rstd_s[L_];
  __shared__ float decay_s, m0_s;

  const int tid  = threadIdx.x;
  const int lane = tid & 31;
  const int wv   = tid >> 5;        // wave id 0..7 (uniform per wave)
  const int l15  = lane & 15;
  const int half = lane >> 4;

  const int bh = blockIdx.x;        // b*NH + h
  const int b  = bh / NH_;
  const int h  = bh % NH_;

  const float* qb = q + (size_t)b * S_ * D_ + (size_t)h * DH_;
  const float* kb = k + (size_t)b * S_ * D_ + (size_t)h * DH_;
  const float* vb = v + (size_t)b * S_ * D_ + (size_t)h * DH_;
  const size_t gbase = (size_t)bh * S_;

  // ---- init state: f32 accumulators + bf16 mirror -------------------------
  v8f Cacc[8];
  {
    const float* c0g = c0_in + (size_t)bh * DH_ * DH_;
#pragma unroll
    for (int j = 0; j < 8; ++j) {
      v8f c;
#pragma unroll
      for (int r = 0; r < 8; ++r) {
        const int d1 = wv * 16 + r + half * 8;
        const int d2 = j * 16 + l15;
        const float cv = c0g[(size_t)d1 * DH_ + d2];
        c[r] = cv;
        Ct_lds[d2 * DH_ + d1] = (bf16)cv;
      }
      Cacc[j] = c;
    }
    if (tid < DH_) n_lds[tid] = n0_in[(size_t)bh * DH_ + tid];
    if (tid == 0)  m0_s = m0_in[bh];
  }
  __syncthreads();

  for (int ch = 0; ch < NCH_; ++ch) {
    const int t0 = ch * L_;

    // ---- s1: stage Q, K(scaled), V^T chunk into LDS as bf16 ---------------
    {
      const int t = tid >> 3, p = tid & 7, d0 = p * 16;
      const float* qp = qb + (size_t)(t0 + t) * D_ + d0;
      const float* kp = kb + (size_t)(t0 + t) * D_ + d0;
      const float* vp = vb + (size_t)(t0 + t) * D_ + d0;
#pragma unroll
      for (int e = 0; e < 16; ++e) {
        Q_lds[t * DH_ + d0 + e]   = (bf16)qp[e];
        K_lds[t * DH_ + d0 + e]   = (bf16)(kp[e] * SCALE_);
        Vt_lds[(d0 + e) * L_ + t] = (bf16)vp[e];
      }
      if (tid < L_) {
        i_s[tid] = ig_ws[gbase + t0 + tid];
        f_s[tid] = fg_ws[gbase + t0 + tid];
      }
      if (ch + 1 < NCH_) {
        __builtin_prefetch((const void*)(qp + (size_t)L_ * D_), 0, 1);
        __builtin_prefetch((const void*)(kp + (size_t)L_ * D_), 0, 1);
        __builtin_prefetch((const void*)(vp + (size_t)L_ * D_), 0, 1);
      }
    }
    __syncthreads();

    // ---- s2: serial gate scan (stabilizer m_t, log-decays) ----------------
    if (tid == 0) {
      float m_prev = m0_s, Ac = 0.f;
      for (int t = 0; t < L_; ++t) {
        const float lf = log_sigmoid_f(f_s[t]);
        Ac += lf;
        A_s[t] = Ac;
        const float mn = fmaxf(lf + m_prev, i_s[t]);
        m_s[t] = mn;
        m_prev = mn;
      }
      const float mL = m_s[L_ - 1], AL = A_s[L_ - 1], m0v = m0_s;
      for (int t = 0; t < L_; ++t) {
        rowinter_s[t] = __expf(A_s[t] + m0v - m_s[t]);
        wstate_s[t]   = __expf(i_s[t] + AL - A_s[t] - mL);
      }
      decay_s = __expf(AL + m0v - mL);
    }
    __syncthreads();

    // ---- s3: S = Q @ K_sc^T (waves 0-3) | build Kw (waves 4-7) ------------
    v8f Sacc;
#pragma unroll
    for (int r = 0; r < 8; ++r) Sacc[r] = 0.f;
    if (wv < 4) {
      const int ti = wv >> 1, ui = wv & 1;
      const bf16* arow = &Q_lds[(ti * 16 + l15) * DH_];
      const bf16* brow = &K_lds[(ui * 16 + l15) * DH_];
#pragma unroll
      for (int kbk = 0; kbk < 4; ++kbk) {
        const v16bf a  = frag_a8(arow + kbk * 32, half);
        const v16bf bm = frag_b16(brow + kbk * 32 + half * 16);
        Sacc = WMMA_BF16(a, bm, Sacc);
      }
    } else {
      const int d = tid - 128;            // 0..127
#pragma unroll
      for (int t = 0; t < L_; ++t)
        Kw_lds[d * L_ + t] = (bf16)((float)K_lds[t * DH_ + d] * wstate_s[t]);
    }
    __syncthreads();

    // ---- s4: causal mask + decay onto S (waves 0-3) | q.n0 (waves 4-7) ----
    if (wv < 4) {
      const int ti = wv >> 1, ui = wv & 1;
#pragma unroll
      for (int r = 0; r < 8; ++r) {
        const int t = ti * 16 + r + half * 8;
        const int u = ui * 16 + l15;
        float val = 0.f;
        if (u <= t)
          val = Sacc[r] * __expf(i_s[u] - A_s[u] + A_s[t] - m_s[t]);
        SD_lds[t * L_ + u] = (bf16)val;
      }
    } else {
      const int loc = tid - 128;
      const int t = loc >> 2, pp = loc & 3;
      float sum = 0.f;
      for (int d = pp * 32; d < pp * 32 + 32; ++d)
        sum += (float)Q_lds[t * DH_ + d] * n_lds[d];
      qred[t][pp] = sum;
    }
    __syncthreads();

    // ---- s5: denominators | n-state update --------------------------------
    if (tid < L_) {
      const int t = tid;
      const float qdn = qred[t][0] + qred[t][1] + qred[t][2] + qred[t][3];
      float rs = 0.f;
      for (int u = 0; u <= t; ++u) rs += (float)SD_lds[t * L_ + u];
      const float qn = rowinter_s[t] * qdn + rs;
      denom_s[t] = fmaxf(fabsf(qn), __expf(-m_s[t])) + EPS_;
      if (t == 0) m0_s = m_s[L_ - 1];   // carry stabilizer to next chunk
    } else if (tid < 32 + DH_) {
      const int d = tid - 32;
      float sum = 0.f;
#pragma unroll
      for (int t = 0; t < L_; ++t) sum += (float)Kw_lds[d * L_ + t];
      n_lds[d] = decay_s * n_lds[d] + sum;
    }
    __syncthreads();

    // ---- s6: H = diag(binter) * (Q@C0) + (D.S)@V, then /= denom -----------
#pragma unroll
    for (int ii = 0; ii < 2; ++ii) {
      const int tI = wv * 2 + ii;         // 0..15
      const int ti = tI >> 3, j = tI & 7;
      v8f acc;
#pragma unroll
      for (int r = 0; r < 8; ++r) acc[r] = 0.f;
      const bf16* arow = &Q_lds[(ti * 16 + l15) * DH_];
      const bf16* brow = &Ct_lds[(j * 16 + l15) * DH_];
#pragma unroll
      for (int kbk = 0; kbk < 4; ++kbk) {
        const v16bf a  = frag_a8(arow + kbk * 32, half);
        const v16bf bm = frag_b16(brow + kbk * 32 + half * 16);
        acc = WMMA_BF16(a, bm, acc);
      }
#pragma unroll
      for (int r = 0; r < 8; ++r)
        acc[r] *= rowinter_s[ti * 16 + r + half * 8];
      {
        const v16bf a  = frag_a8(&SD_lds[(ti * 16 + l15) * L_], half);
        const v16bf bm = frag_b16(&Vt_lds[(j * 16 + l15) * L_ + half * 16]);
        acc = WMMA_BF16(a, bm, acc);
      }
#pragma unroll
      for (int r = 0; r < 8; ++r) {
        const int t = ti * 16 + r + half * 8;
        Hbuf[t * DH_ + j * 16 + l15] = acc[r] / denom_s[t];
      }
    }
    __syncthreads();

    // ---- s7: C = decay*C + Kw^T @ V  (f32 accum, refresh bf16 mirror) -----
#pragma unroll
    for (int j = 0; j < 8; ++j) {
      v8f c = Cacc[j];
#pragma unroll
      for (int r = 0; r < 8; ++r) c[r] *= decay_s;
      const v16bf a  = frag_a8(&Kw_lds[(wv * 16 + l15) * L_], half);
      const v16bf bm = frag_b16(&Vt_lds[(j * 16 + l15) * L_ + half * 16]);
      c = WMMA_BF16(a, bm, c);
      Cacc[j] = c;
#pragma unroll
      for (int r = 0; r < 8; ++r)
        Ct_lds[(j * 16 + l15) * DH_ + wv * 16 + r + half * 8] = (bf16)c[r];
    }
    __syncthreads();

    // ---- s8-s10: layernorm over DH and store ------------------------------
    {
      const int t = tid >> 3, p = tid & 7;
      float s1 = 0.f, s2 = 0.f;
#pragma unroll
      for (int e = 0; e < 16; ++e) {
        const float hv = Hbuf[t * DH_ + p * 16 + e];
        s1 += hv; s2 += hv * hv;
      }
      lnsum[t][p] = s1; lnsq[t][p] = s2;
    }
    __syncthreads();
    if (tid < L_) {
      float su = 0.f, sq = 0.f;
#pragma unroll
      for (int p = 0; p < 8; ++p) { su += lnsum[tid][p]; sq += lnsq[tid][p]; }
      const float mu  = su * (1.f / DH_);
      const float var = sq * (1.f / DH_) - mu * mu;
      mu_s[tid]   = mu;
      rstd_s[tid] = rsqrtf(var + 1e-6f);
    }
    __syncthreads();
    {
      const int t = tid >> 3, p = tid & 7;
      const float mu = mu_s[t], rs = rstd_s[t];
      float* op = out_h + ((size_t)b * S_ + t0 + t) * D_ + h * DH_ + p * 16;
      const float* lw = ln_w + h * DH_ + p * 16;
#pragma unroll
      for (int e = 0; e < 16; ++e)
        op[e] = (Hbuf[t * DH_ + p * 16 + e] - mu) * rs * lw[e];
    }
    __syncthreads();
  }

  // ---- final states --------------------------------------------------------
  {
    float* cg = out_c + (size_t)bh * DH_ * DH_;
#pragma unroll
    for (int j = 0; j < 8; ++j) {
#pragma unroll
      for (int r = 0; r < 8; ++r) {
        const int d1 = wv * 16 + r + half * 8;
        const int d2 = j * 16 + l15;
        cg[(size_t)d1 * DH_ + d2] = Cacc[j][r];
      }
    }
    if (tid < DH_) out_n[(size_t)bh * DH_ + tid] = n_lds[tid];
    if (tid == 0)  out_m[bh] = m0_s;
  }
}

// ---------------------------------------------------------------------------
extern "C" void kernel_launch(void* const* d_in, const int* in_sizes, int n_in,
                              void* d_out, int out_size, void* d_ws, size_t ws_size,
                              hipStream_t stream) {
  const float* q   = (const float*)d_in[0];
  const float* k   = (const float*)d_in[1];
  const float* v   = (const float*)d_in[2];
  const float* c0  = (const float*)d_in[3];
  const float* n0  = (const float*)d_in[4];
  const float* m0  = (const float*)d_in[5];
  const float* W_i = (const float*)d_in[6];
  const float* b_i = (const float*)d_in[7];
  const float* W_f = (const float*)d_in[8];
  const float* b_f = (const float*)d_in[9];
  const float* lnw = (const float*)d_in[10];

  float* out = (float*)d_out;
  float* out_h = out;                                   // B*S*D
  float* out_c = out_h + (size_t)B_ * S_ * D_;          // B*NH*DH*DH
  float* out_n = out_c + (size_t)B_ * NH_ * DH_ * DH_;  // B*NH*DH
  float* out_m = out_n + (size_t)B_ * NH_ * DH_;        // B*NH

  float* ig_ws = (float*)d_ws;                          // B*NH*S
  float* fg_ws = ig_ws + (size_t)B_ * NH_ * S_;

  mlstm_gates_kernel<<<B_ * S_, 128, 0, stream>>>(q, k, v, W_i, b_i, W_f, b_f,
                                                  ig_ws, fg_ws);
  mlstm_chunk_kernel<<<B_ * NH_, 256, 0, stream>>>(q, k, v, c0, n0, m0, lnw,
                                                   ig_ws, fg_ws,
                                                   out_h, out_c, out_n, out_m);
}
